// SwinBlock_30374008718062
// MI455X (gfx1250) — matrix operand order repaired and checked by reading, not compiled, not run on hardware
//
#include <hip/hip_runtime.h>
#include <hip/hip_bf16.h>
#include <math.h>

// ---------------------------------------------------------------------------
// Swin block (B=32, C=384, H=W=56, ws=7, shift=3, NH=12, HD=32) on gfx1250.
// All GEMMs run on v_wmma_f32_16x16x32_f16, ping-pong double-buffered so
// global_load_b128 clauses overlap the matrix pipe. fp32 accumulate.
// ---------------------------------------------------------------------------

typedef _Float16 half_t;
typedef _Float16 v8h  __attribute__((ext_vector_type(8)));
typedef _Float16 v16h __attribute__((ext_vector_type(16)));
typedef float    v8f  __attribute__((ext_vector_type(8)));

#define TOKS   100352          // 32 * 56 * 56  (= 2048 windows * 49)
#define DIMC   384
#define NHEAD  12
#define HDIM   32
#define NWIN   2048
#define HW     56

__device__ __forceinline__ v16h zero16() {
  v16h z;
#pragma unroll
  for (int i = 0; i < 16; i++) z[i] = (_Float16)0;
  return z;
}

__device__ __forceinline__ v16h cat16(v8h lo, v8h hi) {
  v16h r;
#pragma unroll
  for (int i = 0; i < 8; i++) { r[i] = lo[i]; r[i + 8] = hi[i]; }
  return r;
}

// A fragment (16x32 f16, M x K): lanes 0-15 -> K 0..7 & 16..23 of row `lane`,
// lanes 16-31 -> K 8..15 & 24..31 of row `lane-16`.
__device__ __forceinline__ v16h load_a_frag(const half_t* __restrict__ Arow,
                                            int k0, bool hiHalf) {
  const half_t* p = Arow + k0 + (hiHalf ? 8 : 0);
  return cat16(*(const v8h*)p, *(const v8h*)(p + 16));
}

// B fragment (32x16 f16, K x N), Bcol points at Bt[col][0] (col-contiguous K):
// lanes 0-15 -> K 0..7 & 16..23, lanes 16-31 -> K 8..15 & 24..31.
__device__ __forceinline__ v16h load_b_frag(const half_t* __restrict__ Bcol,
                                            int k0, bool hiHalf) {
  const half_t* p = Bcol + k0 + (hiHalf ? 8 : 0);
  return cat16(*(const v8h*)p, *(const v8h*)(p + 16));
}

__device__ __forceinline__ v8f wmma_f16(v16h a, v16h b, v8f c) {
  return __builtin_amdgcn_wmma_f32_16x16x32_f16(false, a, false, b, (short)0,
                                                c, false, false);
}

// ---------------------------------------------------------------------------
// One wave computes a 32(M) x 64(N) tile: 8 WMMA per 32-wide K step, 12 x16B
// loads, ping-pong double buffered over 64-wide K chunks (K % 64 == 0).
// ---------------------------------------------------------------------------
template <int K>
__device__ __forceinline__ void gemm_core(const half_t* __restrict__ A,
                                          const half_t* __restrict__ Bt,
                                          size_t row_base, size_t col_base,
                                          v8f acc[2][4]) {
  const int lane = threadIdx.x & 31;
  const int l15 = lane & 15;
  const bool hiH = lane >= 16;
  const half_t* Arow0 = A + (row_base + (size_t)l15) * K;
  const half_t* Arow1 = Arow0 + (size_t)16 * K;
  const half_t* Bc = Bt + (col_base + (size_t)l15) * K;

  v16h a0[2], a1[2], b[2][4];

  // prologue: fragment set 0 at k = 0
  a0[0] = load_a_frag(Arow0, 0, hiH);
  a1[0] = load_a_frag(Arow1, 0, hiH);
#pragma unroll
  for (int nt = 0; nt < 4; nt++) b[0][nt] = load_b_frag(Bc + nt * 16 * K, 0, hiH);

#pragma unroll 2
  for (int k0 = 0; k0 < K; k0 += 64) {
    // prefetch set 1 (k0 + 32) while set-0 WMMAs run
    a0[1] = load_a_frag(Arow0, k0 + 32, hiH);
    a1[1] = load_a_frag(Arow1, k0 + 32, hiH);
#pragma unroll
    for (int nt = 0; nt < 4; nt++)
      b[1][nt] = load_b_frag(Bc + nt * 16 * K, k0 + 32, hiH);

#pragma unroll
    for (int nt = 0; nt < 4; nt++) {
      acc[0][nt] = wmma_f16(a0[0], b[0][nt], acc[0][nt]);
      acc[1][nt] = wmma_f16(a1[0], b[0][nt], acc[1][nt]);
    }

    if (k0 + 64 < K) {  // prefetch set 0 (k0 + 64) while set-1 WMMAs run
      a0[0] = load_a_frag(Arow0, k0 + 64, hiH);
      a1[0] = load_a_frag(Arow1, k0 + 64, hiH);
#pragma unroll
      for (int nt = 0; nt < 4; nt++)
        b[0][nt] = load_b_frag(Bc + nt * 16 * K, k0 + 64, hiH);
    }

#pragma unroll
    for (int nt = 0; nt < 4; nt++) {
      acc[0][nt] = wmma_f16(a0[1], b[1][nt], acc[0][nt]);
      acc[1][nt] = wmma_f16(a1[1], b[1][nt], acc[1][nt]);
    }
  }
}

__device__ __forceinline__ void zero_acc(v8f acc[2][4]) {
#pragma unroll
  for (int mt = 0; mt < 2; mt++)
#pragma unroll
    for (int nt = 0; nt < 4; nt++)
#pragma unroll
      for (int r = 0; r < 8; r++) acc[mt][nt][r] = 0.f;
}

// --------------------------- weight packing -------------------------------
// W (K x N f32, row-major) -> Wt (N x K f16): B operand wants contiguous K.
__global__ __launch_bounds__(256)
void pack_wT(const float* __restrict__ W, half_t* __restrict__ Wt, int K, int N) {
  int idx = blockIdx.x * 256 + threadIdx.x;
  if (idx >= K * N) return;
  int n = idx / K, k = idx % K;
  Wt[(size_t)n * K + k] = (half_t)W[(size_t)k * N + n];
}

// ------------------- LN1 + shift + window partition ------------------------
// token t in window order: t = ((b*8+wh)*8+ww)*49 + i*7 + j
__global__ __launch_bounds__(256)
void ln1_kernel(const float* __restrict__ x, const float* __restrict__ g,
                const float* __restrict__ bb, half_t* __restrict__ hout) {
  int wave = threadIdx.x >> 5, lane = threadIdx.x & 31;
  size_t t = (size_t)blockIdx.x * 8 + wave;
  int win = (int)(t / 49), p = (int)(t % 49);
  int b = win >> 6, wim = win & 63;
  int wh = wim >> 3, ww = wim & 7;
  int i = p / 7, j = p % 7;
  int h = wh * 7 + i + 3; if (h >= HW) h -= HW;   // roll(-3) inverse lookup
  int w = ww * 7 + j + 3; if (w >= HW) w -= HW;
  const float* px = x + ((size_t)b * DIMC) * 3136 + h * HW + w;
  float vals[12], s = 0.f, sq = 0.f;
#pragma unroll
  for (int q = 0; q < 12; q++) {
    float v = px[(size_t)(q * 32 + lane) * 3136];
    vals[q] = v; s += v; sq += v * v;
  }
#pragma unroll
  for (int m = 16; m > 0; m >>= 1) {
    s += __shfl_xor(s, m, 32);
    sq += __shfl_xor(sq, m, 32);
  }
  float mu = s * (1.f / DIMC);
  float inv = rsqrtf(sq * (1.f / DIMC) - mu * mu + 1e-5f);
#pragma unroll
  for (int q = 0; q < 12; q++) {
    int c = q * 32 + lane;
    hout[t * DIMC + c] = (half_t)((vals[q] - mu) * inv * g[c] + bb[c]);
  }
}

// LN2 over x1 (NHWC f32, contiguous channels)
__global__ __launch_bounds__(256)
void ln2_kernel(const float* __restrict__ x1, const float* __restrict__ g,
                const float* __restrict__ bb, half_t* __restrict__ hout) {
  int wave = threadIdx.x >> 5, lane = threadIdx.x & 31;
  size_t u = (size_t)blockIdx.x * 8 + wave;
  const float* px = x1 + u * DIMC;
  float vals[12], s = 0.f, sq = 0.f;
#pragma unroll
  for (int q = 0; q < 12; q++) {
    float v = px[q * 32 + lane];
    vals[q] = v; s += v; sq += v * v;
  }
#pragma unroll
  for (int m = 16; m > 0; m >>= 1) {
    s += __shfl_xor(s, m, 32);
    sq += __shfl_xor(sq, m, 32);
  }
  float mu = s * (1.f / DIMC);
  float inv = rsqrtf(sq * (1.f / DIMC) - mu * mu + 1e-5f);
#pragma unroll
  for (int q = 0; q < 12; q++) {
    int c = q * 32 + lane;
    hout[u * DIMC + c] = (half_t)((vals[q] - mu) * inv * g[c] + bb[c]);
  }
}

// ------------------------------ QKV GEMM -----------------------------------
__global__ __launch_bounds__(256)
void gemm_qkv(const half_t* __restrict__ A, const half_t* __restrict__ Bt,
              const float* __restrict__ bias, half_t* __restrict__ qb,
              half_t* __restrict__ kb, half_t* __restrict__ vb) {
  const int wave = threadIdx.x >> 5;
  size_t row_base = (size_t)blockIdx.y * 256 + wave * 32;
  size_t col_base = (size_t)blockIdx.x * 64;
  v8f acc[2][4];
  zero_acc(acc);
  gemm_core<DIMC>(A, Bt, row_base, col_base, acc);

  const int lane = threadIdx.x & 31, l15 = lane & 15;
  const int roff = (lane >= 16) ? 8 : 0;
  const float scale = 0.17677669529663687f;  // 32^-0.5 folded into q
  int sgrp = (int)col_base / DIMC;           // 64-col block never straddles q/k/v
  half_t* dst = (sgrp == 0) ? qb : ((sgrp == 1) ? kb : vb);
  int cbase = (int)col_base - sgrp * DIMC;
#pragma unroll
  for (int mt = 0; mt < 2; mt++)
#pragma unroll
    for (int nt = 0; nt < 4; nt++)
#pragma unroll
      for (int r = 0; r < 8; r++) {
        size_t row = row_base + mt * 16 + r + roff;
        int col = (int)col_base + nt * 16 + l15;
        float v = acc[mt][nt][r] + bias[col];
        if (sgrp == 0) v *= scale;
        dst[row * DIMC + (cbase + nt * 16 + l15)] = (half_t)v;
      }
}

// ------------------------------ attention ----------------------------------
// One block per (head, window). N=49 padded to 64. 4 waves.
__global__ __launch_bounds__(128)
void attn_kernel(const half_t* __restrict__ qb, const half_t* __restrict__ kb,
                 const half_t* __restrict__ vb, const float* __restrict__ rpb,
                 half_t* __restrict__ ob) {
  __shared__ __align__(16) float  S[64 * 64];
  __shared__ __align__(16) half_t P[64 * 64];
  __shared__ __align__(16) half_t Vt[32 * 64];

  const int head = blockIdx.x;
  const int win = blockIdx.y;
  const int tid = threadIdx.x;
  const int wave = tid >> 5, lane = tid & 31;
  const int l15 = lane & 15;
  const bool hiH = lane >= 16;
  const size_t base = (size_t)win * 49 * DIMC + head * HDIM;

  // ---- S = (q*scale) @ k^T : K = HDIM = 32, exactly one WMMA K-step ----
  {
    int m0 = wave * 16;
    int arow = m0 + l15; if (arow > 48) arow = 48;       // clamp pad rows
    v16h a = load_a_frag(qb + base + (size_t)arow * DIMC, 0, hiH);
    v8f acc[4];
#pragma unroll
    for (int nt = 0; nt < 4; nt++) {
#pragma unroll
      for (int r = 0; r < 8; r++) acc[nt][r] = 0.f;
      int tokc = nt * 16 + l15;
      v16h b = zero16();
      if (tokc < 49) b = load_b_frag(kb + base + (size_t)tokc * DIMC, 0, hiH);
      acc[nt] = wmma_f16(a, b, acc[nt]);
    }
    int roff = hiH ? 8 : 0;
#pragma unroll
    for (int nt = 0; nt < 4; nt++)
#pragma unroll
      for (int r = 0; r < 8; r++)
        S[(m0 + r + roff) * 64 + nt * 16 + l15] = acc[nt][r];
  }

  // ---- stage V transposed: Vt[hd][tok], pad tok 49..63 with zeros ----
  for (int idx = tid; idx < 32 * 64; idx += 128) {
    int hd = idx >> 6, tok = idx & 63;
    Vt[hd * 64 + tok] =
        (tok < 49) ? vb[base + (size_t)tok * DIMC + hd] : (half_t)0;
  }
  __syncthreads();

  // ---- rel-pos bias + shift mask + softmax (one thread per row) ----
  if (tid < 49) {
    int row = tid;
    int wim = win & 63, wh = wim >> 3, ww = wim & 7;
    int i1 = row / 7, j1 = row % 7;
    int hs1 = wh * 7 + i1, ws1 = ww * 7 + j1;
    int lab1 = ((hs1 < 49) ? 0 : ((hs1 < 53) ? 1 : 2)) * 3 +
               ((ws1 < 49) ? 0 : ((ws1 < 53) ? 1 : 2));
    float mx = -1e30f;
    for (int m = 0; m < 49; m++) {
      int i2 = m / 7, j2 = m % 7;
      int ridx = (i1 - i2 + 6) * 13 + (j1 - j2 + 6);
      float s = S[row * 64 + m] + rpb[ridx * NHEAD + head];
      int hs2 = wh * 7 + i2, ws2 = ww * 7 + j2;
      int lab2 = ((hs2 < 49) ? 0 : ((hs2 < 53) ? 1 : 2)) * 3 +
                 ((ws2 < 49) ? 0 : ((ws2 < 53) ? 1 : 2));
      if (lab2 != lab1) s -= 100.0f;
      S[row * 64 + m] = s;
      mx = fmaxf(mx, s);
    }
    float sum = 0.f;
    for (int m = 0; m < 49; m++) {
      float e = __expf(S[row * 64 + m] - mx);
      S[row * 64 + m] = e;
      sum += e;
    }
    float inv = 1.0f / sum;
    for (int m = 0; m < 49; m++) P[row * 64 + m] = (half_t)(S[row * 64 + m] * inv);
    for (int m = 49; m < 64; m++) P[row * 64 + m] = (half_t)0;  // K padding -> 0
  }
  __syncthreads();

  // ---- O = P @ V : K = 64 (2 WMMA K-steps), N = 32 (2 tiles) ----
  {
    int m0 = wave * 16;
    v8f acc[2];
#pragma unroll
    for (int nt = 0; nt < 2; nt++)
#pragma unroll
      for (int r = 0; r < 8; r++) acc[nt][r] = 0.f;
    for (int k0 = 0; k0 < 64; k0 += 32) {
      v16h a = load_a_frag(&P[(m0 + l15) * 64], k0, hiH);
#pragma unroll
      for (int nt = 0; nt < 2; nt++) {
        v16h b = load_b_frag(&Vt[(nt * 16 + l15) * 64], k0, hiH);
        acc[nt] = wmma_f16(a, b, acc[nt]);
      }
    }
    int roff = hiH ? 8 : 0;
#pragma unroll
    for (int nt = 0; nt < 2; nt++)
#pragma unroll
      for (int r = 0; r < 8; r++) {
        int tok = m0 + r + roff;
        if (tok < 49)
          ob[base + (size_t)tok * DIMC + nt * 16 + l15] = (half_t)acc[nt][r];
      }
  }
}

// --------------------- proj GEMM + residual + reverse ----------------------
__global__ __launch_bounds__(256)
void gemm_proj(const half_t* __restrict__ A, const half_t* __restrict__ Bt,
               const float* __restrict__ bias, const float* __restrict__ x,
               float* __restrict__ x1) {
  const int wave = threadIdx.x >> 5;
  size_t row_base = (size_t)blockIdx.y * 256 + wave * 32;
  size_t col_base = (size_t)blockIdx.x * 64;
  v8f acc[2][4];
  zero_acc(acc);
  gemm_core<DIMC>(A, Bt, row_base, col_base, acc);

  const int lane = threadIdx.x & 31, l15 = lane & 15;
  const int roff = (lane >= 16) ? 8 : 0;
#pragma unroll
  for (int mt = 0; mt < 2; mt++)
#pragma unroll
    for (int nt = 0; nt < 4; nt++)
#pragma unroll
      for (int r = 0; r < 8; r++) {
        size_t t = row_base + mt * 16 + r + roff;
        int col = (int)col_base + nt * 16 + l15;
        int win = (int)(t / 49), p = (int)(t % 49);
        int b = win >> 6, wim = win & 63;
        int wh = wim >> 3, ww = wim & 7;
        int i = p / 7, j = p % 7;
        int h = wh * 7 + i + 3; if (h >= HW) h -= HW;    // roll(+3)
        int w = ww * 7 + j + 3; if (w >= HW) w -= HW;
        float sc = x[((size_t)b * DIMC + col) * 3136 + h * HW + w];
        x1[((size_t)b * 3136 + h * HW + w) * DIMC + col] =
            sc + acc[mt][nt][r] + bias[col];
      }
}

// ------------------------------ FC1 + GELU ---------------------------------
__global__ __launch_bounds__(256)
void gemm_fc1(const half_t* __restrict__ A, const half_t* __restrict__ Bt,
              const float* __restrict__ bias, half_t* __restrict__ mid) {
  const int wave = threadIdx.x >> 5;
  size_t row_base = (size_t)blockIdx.y * 256 + wave * 32;
  size_t col_base = (size_t)blockIdx.x * 64;
  v8f acc[2][4];
  zero_acc(acc);
  gemm_core<DIMC>(A, Bt, row_base, col_base, acc);

  const int lane = threadIdx.x & 31, l15 = lane & 15;
  const int roff = (lane >= 16) ? 8 : 0;
#pragma unroll
  for (int mt = 0; mt < 2; mt++)
#pragma unroll
    for (int nt = 0; nt < 4; nt++)
#pragma unroll
      for (int r = 0; r < 8; r++) {
        size_t row = row_base + mt * 16 + r + roff;
        int col = (int)col_base + nt * 16 + l15;
        float v = acc[mt][nt][r] + bias[col];
        v = 0.5f * v * (1.0f + erff(v * 0.70710678118654752f));  // exact GELU
        mid[row * 1536 + col] = (half_t)v;
      }
}

// --------------------- FC2 + residual + NCHW store -------------------------
__global__ __launch_bounds__(256)
void gemm_fc2(const half_t* __restrict__ A, const half_t* __restrict__ Bt,
              const float* __restrict__ bias, const float* __restrict__ x1,
              float* __restrict__ out) {
  const int wave = threadIdx.x >> 5;
  size_t row_base = (size_t)blockIdx.y * 256 + wave * 32;
  size_t col_base = (size_t)blockIdx.x * 64;
  v8f acc[2][4];
  zero_acc(acc);
  gemm_core<1536>(A, Bt, row_base, col_base, acc);

  const int lane = threadIdx.x & 31, l15 = lane & 15;
  const int roff = (lane >= 16) ? 8 : 0;
#pragma unroll
  for (int mt = 0; mt < 2; mt++)
#pragma unroll
    for (int nt = 0; nt < 4; nt++)
#pragma unroll
      for (int r = 0; r < 8; r++) {
        size_t u = row_base + mt * 16 + r + roff;  // NHWC token
        int col = (int)col_base + nt * 16 + l15;
        float v = acc[mt][nt][r] + bias[col] + x1[u * DIMC + col];
        int b = (int)(u / 3136), pix = (int)(u % 3136);
        out[((size_t)b * DIMC + col) * 3136 + pix] = v;
      }
}

// ---------------------------------------------------------------------------
extern "C" void kernel_launch(void* const* d_in, const int* in_sizes, int n_in,
                              void* d_out, int out_size, void* d_ws, size_t ws_size,
                              hipStream_t stream) {
  (void)in_sizes; (void)n_in; (void)out_size; (void)ws_size;
  const float* x      = (const float*)d_in[0];
  const float* ln1_g  = (const float*)d_in[1];
  const float* ln1_b  = (const float*)d_in[2];
  const float* qkv_w  = (const float*)d_in[3];
  const float* qkv_b  = (const float*)d_in[4];
  const float* rpb    = (const float*)d_in[5];
  const float* proj_w = (const float*)d_in[6];
  const float* proj_b = (const float*)d_in[7];
  const float* ln2_g  = (const float*)d_in[8];
  const float* ln2_b  = (const float*)d_in[9];
  const float* pw1_w  = (const float*)d_in[10];
  const float* pw1_b  = (const float*)d_in[11];
  const float* pw2_w  = (const float*)d_in[12];
  const float* pw2_b  = (const float*)d_in[13];

  char* ws = (char*)d_ws;
  const size_t bh = (size_t)TOKS * DIMC * sizeof(half_t);   // 77,070,336 B
  half_t* hbuf  = (half_t*)ws;                 // LN1 out (window order); reused as LN2 out
  half_t* qb    = (half_t*)(ws + 1 * bh);
  half_t* kb    = (half_t*)(ws + 2 * bh);
  half_t* vb    = (half_t*)(ws + 3 * bh);
  half_t* ob    = (half_t*)(ws + 4 * bh);
  float*  x1    = (float*)(ws + 5 * bh);       // NHWC f32, TOKS*384*4
  size_t off_w  = 5 * bh + (size_t)TOKS * DIMC * sizeof(float);
  half_t* qkvT  = (half_t*)(ws + off_w);
  half_t* projT = qkvT + (size_t)1152 * DIMC;
  half_t* pw1T  = projT + (size_t)DIMC * DIMC;
  half_t* pw2T  = pw1T + (size_t)1536 * DIMC;
  half_t* mid   = qb;     // q..o region = TOKS*1536*2 bytes exactly, dead by FC1
  half_t* h2    = hbuf;   // dead after QKV

  // pack weights -> f16 transposed
  int n;
  n = DIMC * 1152; pack_wT<<<(n + 255) / 256, 256, 0, stream>>>(qkv_w, qkvT, DIMC, 1152);
  n = DIMC * DIMC; pack_wT<<<(n + 255) / 256, 256, 0, stream>>>(proj_w, projT, DIMC, DIMC);
  n = DIMC * 1536; pack_wT<<<(n + 255) / 256, 256, 0, stream>>>(pw1_w, pw1T, DIMC, 1536);
  n = 1536 * DIMC; pack_wT<<<(n + 255) / 256, 256, 0, stream>>>(pw2_w, pw2T, 1536, DIMC);

  ln1_kernel<<<TOKS / 8, 256, 0, stream>>>(x, ln1_g, ln1_b, hbuf);
  gemm_qkv<<<dim3(18, 392), 256, 0, stream>>>(hbuf, qkvT, qkv_b, qb, kb, vb);
  attn_kernel<<<dim3(NHEAD, NWIN), 128, 0, stream>>>(qb, kb, vb, rpb, ob);
  gemm_proj<<<dim3(6, 392), 256, 0, stream>>>(ob, projT, proj_b, x, x1);
  ln2_kernel<<<TOKS / 8, 256, 0, stream>>>(x1, ln2_g, ln2_b, h2);
  gemm_fc1<<<dim3(24, 392), 256, 0, stream>>>(h2, pw1T, pw1_b, mid);
  gemm_fc2<<<dim3(6, 392), 256, 0, stream>>>(mid, pw2T, pw2_b, x1, (float*)d_out);
}
